// UnifiedQuanvolutionClassifier_65481071407222
// MI455X (gfx1250) — compile-verified, matrix operand order checked by reading, not run on hardware
//
#include <hip/hip_runtime.h>
#include <hip/hip_bf16.h>
#include <math.h>

// ---------------------------------------------------------------------------
// Quanvolution classifier, MI455X (gfx1250, wave32, WMMA) — round 3.
//
//   K1 quanv:  Phi^T = U x Psi^T (2 WMMA), Probs=|Phi|^2, Z = Signs x Probs^T
//              (1 WMMA).  Stores feats as f16x4 per patch (8B store),
//              rows padded 784 -> 800.
//   K2 prep:   Wh[16][800] = zero-padded f16 copy of W; zero feats row tails.
//   K3 head:   logits = feats @ W^T as WMMA: 16 images x 16 classes per wave,
//              25 chained v_wmma_f32_16x16x32_f16 over K=800.  No shuffles.
//   K4 smax:   one thread per image: +b, log_softmax over 10 classes.
//
// All cross-lane data movement eliminated (ds = 0); matrix math is 100% WMMA.
// ---------------------------------------------------------------------------

typedef __attribute__((ext_vector_type(16))) _Float16 v16h;
typedef __attribute__((ext_vector_type(8)))  _Float16 v8h;
typedef __attribute__((ext_vector_type(4)))  _Float16 v4h;
typedef __attribute__((ext_vector_type(8)))  float    v8f;

#define NIMG   4096
#define NPATCH 196
#define PTOT   (NIMG * NPATCH)          // 802816 patches
#define TILES  (PTOT / 16)              // 50176 wave-tiles (exact)
#define WPB    8                        // waves per block (256 threads)
#define QBLK   (TILES / WPB)            // 6272 blocks (exact)
#define FROW   800                      // feats row length (784 + 16 pad), 25*32
#define HTILE  (NIMG / 16)              // 256 head tiles
#define HBLK   (HTILE / WPB)            // 32 head blocks

// ---------------- K1: quantum layer (3 WMMAs per 16-patch tile) ------------
__global__ __launch_bounds__(256)
void quanv_wmma_kernel(const float* __restrict__ x,     // [4096*784]
                       const float* __restrict__ U_re,  // [16*16]
                       const float* __restrict__ U_im,  // [16*16]
                       _Float16* __restrict__ feats)    // [NIMG*FROW] f16
{
    const int lane = threadIdx.x & 31;
    const int wave = threadIdx.x >> 5;
    const int tile = blockIdx.x * WPB + wave;

    const int n = lane & 15;   // A: U row i / B,C: patch column
    const int h = lane >> 4;   // half-wave: K-split (states 8h..8h+7)

    // A matrices: A[i,k] = U[i,k]; lane row i = n, halves 0..7 -> K = 8h+s.
    v16h a_re = {};
    v16h a_im = {};
    {
        const float4* ur = reinterpret_cast<const float4*>(U_re + n * 16 + 8 * h);
        const float4* ui = reinterpret_cast<const float4*>(U_im + n * 16 + 8 * h);
        const float4 r0 = ur[0], r1 = ur[1];
        const float4 i0 = ui[0], i1 = ui[1];
        a_re[0] = (_Float16)r0.x; a_re[1] = (_Float16)r0.y;
        a_re[2] = (_Float16)r0.z; a_re[3] = (_Float16)r0.w;
        a_re[4] = (_Float16)r1.x; a_re[5] = (_Float16)r1.y;
        a_re[6] = (_Float16)r1.z; a_re[7] = (_Float16)r1.w;
        a_im[0] = (_Float16)i0.x; a_im[1] = (_Float16)i0.y;
        a_im[2] = (_Float16)i0.z; a_im[3] = (_Float16)i0.w;
        a_im[4] = (_Float16)i1.x; a_im[5] = (_Float16)i1.y;
        a_im[6] = (_Float16)i1.z; a_im[7] = (_Float16)i1.w;
    }

    // B matrix: B[k,p] = psi[patch p, state k]; lane = patch column.
    const int pglob = tile * 16 + n;
    const int img   = pglob / NPATCH;
    const int pp    = pglob - img * NPATCH;
    const int py    = pp / 14;
    const int px    = pp - py * 14;
    const float* xp = x + img * 784 + (2 * py) * 28 + 2 * px;

    const float2 row0 = *reinterpret_cast<const float2*>(xp);       // dy=0
    const float2 row1 = *reinterpret_cast<const float2*>(xp + 28);  // dy=1
    const float t0 = 0.5f * row0.x;   // wire order = (dy,dx) row-major
    const float t1 = 0.5f * row0.y;
    const float t2 = 0.5f * row1.x;
    const float t3 = 0.5f * row1.y;

    // state k = b0*8+b1*4+b2*2+b3, wire0 = MSB.  k = 8h + s => b0 = h.
    const float v0 = h ? __sinf(t0) : __cosf(t0);
    const float c1 = __cosf(t1), s1 = __sinf(t1);
    const float c2 = __cosf(t2), s2 = __sinf(t2);
    const float c3 = __cosf(t3), s3 = __sinf(t3);

    v16h bp = {};
#pragma unroll
    for (int s = 0; s < 8; ++s) {
        const float amp = v0 * ((s & 4) ? s1 : c1)
                             * ((s & 2) ? s2 : c2)
                             * ((s & 1) ? s3 : c3);
        bp[s] = (_Float16)amp;                  // halves 8..15 zero (K pad)
    }

    // WMMA 1/2: Phi^T = U x Psi^T (f32 accumulate).
    v8f phr = {};
    v8f phi = {};
    phr = __builtin_amdgcn_wmma_f32_16x16x32_f16(false, a_re, false, bp,
                                                 (short)0, phr, false, false);
    phi = __builtin_amdgcn_wmma_f32_16x16x32_f16(false, a_im, false, bp,
                                                 (short)0, phi, false, false);

    // probs in C-layout == B-layout of next WMMA: pack to f16 halves.
    v16h pb = {};
#pragma unroll
    for (int r = 0; r < 8; ++r) {
        const float pr = phr[r] * phr[r] + phi[r] * phi[r];
        pb[r] = (_Float16)pr;
    }

    // Sign matrix A[w,k] = 1 - 2*bit_{3-w}(k) for w<4, else 0 (constants).
    v16h sa = {};
    {
        const int sh = (n < 4) ? (3 - n) : 0;
#pragma unroll
        for (int s = 0; s < 8; ++s) {
            const int k = 8 * h + s;
            const float sv = (n < 4) ? (((k >> sh) & 1) ? -1.0f : 1.0f) : 0.0f;
            sa[s] = (_Float16)sv;
        }
    }

    // WMMA 3: Z = Signs x Probs^T.
    v8f z = {};
    z = __builtin_amdgcn_wmma_f32_16x16x32_f16(false, sa, false, pb,
                                               (short)0, z, false, false);

    // Lanes 0-15: z[patch n][wires 0..3] in c[0..3] -> one 8-byte f16x4 store.
    if (h == 0) {
        v4h zz;
        zz[0] = (_Float16)z[0]; zz[1] = (_Float16)z[1];
        zz[2] = (_Float16)z[2]; zz[3] = (_Float16)z[3];
        *reinterpret_cast<v4h*>(feats + img * FROW + pp * 4) = zz;
    }
}

// ---------------- K2: pad W -> f16 Wh[16][800]; zero feats row tails -------
__global__ __launch_bounds__(256)
void prep_kernel(const float* __restrict__ W,       // [10*784]
                 _Float16* __restrict__ Wh,         // [16*FROW]
                 _Float16* __restrict__ feats)      // [NIMG*FROW]
{
    const int idx = blockIdx.x * 256 + threadIdx.x;
    if (idx < 16 * FROW) {
        const int c = idx / FROW;
        const int k = idx - c * FROW;
        Wh[idx] = (c < 10 && k < 784) ? (_Float16)W[c * 784 + k]
                                      : (_Float16)0.0f;
    }
    const int j = idx - 16 * FROW;
    if (j >= 0 && j < NIMG * 16) {
        const int img = j >> 4;
        const int t   = j & 15;
        feats[img * FROW + 784 + t] = (_Float16)0.0f;
    }
}
#define PREP_ELEMS (16 * FROW + NIMG * 16)          // 78336
#define PREP_BLK   (PREP_ELEMS / 256)               // 306 (exact)

// ---------------- K3: head GEMM via chained WMMA ---------------------------
__global__ __launch_bounds__(256)
void head_wmma_kernel(const _Float16* __restrict__ feats,  // [NIMG*FROW]
                      const _Float16* __restrict__ Wh,     // [16*FROW]
                      float* __restrict__ logits)          // [NIMG*16]
{
    const int lane = threadIdx.x & 31;
    const int wave = threadIdx.x >> 5;
    const int tile = blockIdx.x * WPB + wave;   // 16 images per tile

    const int n = lane & 15;                    // A row = image / B,C col = class
    const int h = lane >> 4;

    const _Float16* fa = feats + (tile * 16 + n) * FROW + 8 * h;
    const _Float16* wb = Wh + n * FROW + 8 * h;

    v8f acc = {};
#pragma unroll 5
    for (int kb = 0; kb < FROW; kb += 32) {
        const v8h alo = *reinterpret_cast<const v8h*>(fa + kb);       // K 8h+0..7
        const v8h ahi = *reinterpret_cast<const v8h*>(fa + kb + 16);  // K 16+8h+..
        const v8h blo = *reinterpret_cast<const v8h*>(wb + kb);
        const v8h bhi = *reinterpret_cast<const v8h*>(wb + kb + 16);
        v16h A, B;
#pragma unroll
        for (int i = 0; i < 8; ++i) {
            A[i] = alo[i]; A[8 + i] = ahi[i];
            B[i] = blo[i]; B[8 + i] = bhi[i];
        }
        acc = __builtin_amdgcn_wmma_f32_16x16x32_f16(false, A, false, B,
                                                     (short)0, acc, false, false);
    }

    // C-layout: lane = class n, VGPR r = image row r + 8h.
#pragma unroll
    for (int r = 0; r < 8; ++r)
        logits[(tile * 16 + r + 8 * h) * 16 + n] = acc[r];
}

// ---------------- K4: bias + log_softmax, one thread per image -------------
__global__ __launch_bounds__(256)
void logsoftmax_kernel(const float* __restrict__ logits,  // [NIMG*16]
                       const float* __restrict__ b,       // [10]
                       float* __restrict__ out)           // [NIMG*10]
{
    const int img = blockIdx.x * 256 + threadIdx.x;
    if (img >= NIMG) return;

    float l[10];
#pragma unroll
    for (int c = 0; c < 10; ++c) l[c] = logits[img * 16 + c] + b[c];

    float m = l[0];
#pragma unroll
    for (int c = 1; c < 10; ++c) m = fmaxf(m, l[c]);
    float sum = 0.0f;
#pragma unroll
    for (int c = 0; c < 10; ++c) sum += __expf(l[c] - m);
    const float lse = __logf(sum);
#pragma unroll
    for (int c = 0; c < 10; ++c) out[img * 10 + c] = l[c] - m - lse;
}

// ---------------------------------------------------------------------------
extern "C" void kernel_launch(void* const* d_in, const int* in_sizes, int n_in,
                              void* d_out, int out_size, void* d_ws, size_t ws_size,
                              hipStream_t stream)
{
    const float* x    = (const float*)d_in[0];   // [4096,1,28,28]
    const float* U_re = (const float*)d_in[1];   // [16,16]
    const float* U_im = (const float*)d_in[2];   // [16,16]
    const float* W    = (const float*)d_in[3];   // [10,784]
    const float* b    = (const float*)d_in[4];   // [10]
    float* out        = (float*)d_out;           // [4096,10] log-softmax
    (void)in_sizes; (void)n_in; (void)out_size; (void)ws_size;

    // Workspace layout (all rewritten every call):
    //   feats  f16 [4096*800]          6,553,600 B
    //   Wh     f16 [16*800]               25,600 B
    //   logits f32 [4096*16]             262,144 B        total < 7 MB
    char* ws = (char*)d_ws;
    _Float16* feats  = (_Float16*)ws;
    _Float16* Wh     = (_Float16*)(ws + (size_t)NIMG * FROW * 2);
    float*    logits = (float*)   (ws + (size_t)NIMG * FROW * 2 + 16 * FROW * 2);

    // K1 + K2 are independent; K3 needs both; K4 needs K3.  Stream-ordered.
    quanv_wmma_kernel<<<QBLK, 256, 0, stream>>>(x, U_re, U_im, feats);
    prep_kernel<<<PREP_BLK, 256, 0, stream>>>(W, Wh, feats);
    head_wmma_kernel<<<HBLK, 256, 0, stream>>>(feats, Wh, logits);
    logsoftmax_kernel<<<NIMG / 256, 256, 0, stream>>>(logits, b, out);
}
